// KalmanNet_36953898614889
// MI455X (gfx1250) — matrix operand order repaired and checked by reading, not compiled, not run on hardware
//
#include <hip/hip_runtime.h>
#include <hip/hip_bf16.h>

// ---------------------------------------------------------------------------
// KalmanNet persistent-workgroup implementation for gfx1250 (MI455X).
//  * 8 workgroups x 512 threads (16 wave32s). Each WG owns a batch tile of 16
//    for the full T=256 recurrence; all recurrent state lives in LDS.
//  * All FC/GRU matmuls with K>=32 run on v_wmma_f32_16x16x32_bf16.
//  * WMMA A-operands are kept in bf16 LDS buffers so a fragment build is two
//    pure ds_load_b128s (no conversion VALU). fp32 -> bf16 happens once per
//    element at production time via native __bf16 conversion.
//  * Weights are packed once per launch into bf16 WMMA B-fragment layout
//    (lane&15 -> N, elements + lane half -> K), zero-padded to K%32==0.
// ---------------------------------------------------------------------------

typedef __attribute__((ext_vector_type(16))) __bf16          v16bf;
typedef __attribute__((ext_vector_type(16))) unsigned short  v16u;
typedef __attribute__((ext_vector_type(8)))  unsigned short  v8u;
typedef __attribute__((ext_vector_type(8)))  float           v8f;

#define BLOCK  512
#define NWAVES 16
#define TSTEPS 256

__device__ inline unsigned short f2bf(float f) {
    __bf16 b = (__bf16)f;                      // native cvt, RNE
    return __builtin_bit_cast(unsigned short, b);
}

__device__ inline float sigm(float x) {
    return __fdividef(1.f, 1.f + __expf(-x));
}
__device__ inline float tanh_fast(float x) {
    return 1.f - __fdividef(2.f, __expf(2.f * x) + 1.f);
}

// --- A-fragment loader from bf16 LDS buffer (ISA 7.12.2 16-bit A 16x32) -----
// lane L: M = L&15 ; fragment halves e=0..7 -> K = kb+e, e=8..15 -> K = kb+16+e-8
// where kb = k0 + 8*(L>>4).  => two contiguous 16-byte LDS loads, no VALU.
__device__ inline v16bf a_frag_bf(const unsigned short* A, int stride, int k0, int lane) {
    const int m  = lane & 15;
    const int kb = k0 + ((lane >> 4) << 3);
    const unsigned short* base = A + m * stride + kb;
    v8u lo = *(const v8u*)(base);
    v8u hi = *(const v8u*)(base + 16);
    v16u r;
#pragma unroll
    for (int i = 0; i < 8; ++i) { r[i] = lo[i]; r[8 + i] = hi[i]; }
    return __builtin_bit_cast(v16bf, r);
}

// --- generic WMMA FC: out_f32[16][N] = act( A_bf16[16][K] * Wp + bias ) -----
template <int KT, bool RELU>
__device__ inline void wmma_fc(const unsigned short* A, int a_stride,
                               const unsigned short* Wp, const float* bias,
                               float* out, int out_stride, int NT,
                               int wave, int lane) {
    v16bf afr[KT];
#pragma unroll
    for (int kt = 0; kt < KT; ++kt) afr[kt] = a_frag_bf(A, a_stride, kt * 32, lane);
    for (int nt = wave; nt < NT; nt += NWAVES) {
        v8f acc = {};
        const unsigned short* bp = Wp + (((size_t)nt * KT) << 9) + (lane << 4);
#pragma unroll
        for (int kt = 0; kt < KT; ++kt) {
            v16bf b = __builtin_bit_cast(v16bf, *(const v16u*)bp);
            bp += 512;
            acc = __builtin_amdgcn_wmma_f32_16x16x32_bf16(false, afr[kt], false, b,
                                                          (short)0, acc, false, false);
        }
        const int col = (nt << 4) + (lane & 15);
        const float bb = bias[col];
#pragma unroll
        for (int i = 0; i < 8; ++i) {
            const int row = i + ((lane >> 4) << 3);
            float f = acc[i] + bb;
            if (RELU) f = fmaxf(f, 0.f);
            out[row * out_stride + col] = f;
        }
    }
}

// variant with bf16 (ushort) output (feeds a later WMMA A operand)
template <int KT, bool RELU>
__device__ inline void wmma_fc_bf16out(const unsigned short* A, int a_stride,
                                       const unsigned short* Wp, const float* bias,
                                       unsigned short* out, int out_stride, int NT,
                                       int wave, int lane) {
    v16bf afr[KT];
#pragma unroll
    for (int kt = 0; kt < KT; ++kt) afr[kt] = a_frag_bf(A, a_stride, kt * 32, lane);
    for (int nt = wave; nt < NT; nt += NWAVES) {
        v8f acc = {};
        const unsigned short* bp = Wp + (((size_t)nt * KT) << 9) + (lane << 4);
#pragma unroll
        for (int kt = 0; kt < KT; ++kt) {
            v16bf b = __builtin_bit_cast(v16bf, *(const v16u*)bp);
            bp += 512;
            acc = __builtin_amdgcn_wmma_f32_16x16x32_bf16(false, afr[kt], false, b,
                                                          (short)0, acc, false, false);
        }
        const int col = (nt << 4) + (lane & 15);
        const float bb = bias[col];
#pragma unroll
        for (int i = 0; i < 8; ++i) {
            const int row = i + ((lane >> 4) << 3);
            float f = acc[i] + bb;
            if (RELU) f = fmaxf(f, 0.f);
            out[row * out_stride + col] = f2bf(f);
        }
    }
}

// dual-destination variant (fp32 state + bf16 A-buffer), relu, for o4 -> hSig
template <int KT>
__device__ inline void wmma_fc_dual(const unsigned short* A, int a_stride,
                                    const unsigned short* Wp, const float* bias,
                                    float* outf, int f_stride,
                                    unsigned short* outb, int b_stride, int NT,
                                    int wave, int lane) {
    v16bf afr[KT];
#pragma unroll
    for (int kt = 0; kt < KT; ++kt) afr[kt] = a_frag_bf(A, a_stride, kt * 32, lane);
    for (int nt = wave; nt < NT; nt += NWAVES) {
        v8f acc = {};
        const unsigned short* bp = Wp + (((size_t)nt * KT) << 9) + (lane << 4);
#pragma unroll
        for (int kt = 0; kt < KT; ++kt) {
            v16bf b = __builtin_bit_cast(v16bf, *(const v16u*)bp);
            bp += 512;
            acc = __builtin_amdgcn_wmma_f32_16x16x32_bf16(false, afr[kt], false, b,
                                                          (short)0, acc, false, false);
        }
        const int col = (nt << 4) + (lane & 15);
        const float bb = bias[col];
#pragma unroll
        for (int i = 0; i < 8; ++i) {
            const int row = i + ((lane >> 4) << 3);
            const float f = fmaxf(acc[i] + bb, 0.f);
            outf[row * f_stride + col] = f;
            outb[row * b_stride + col] = f2bf(f);
        }
    }
}

// --- GRU elementwise (biases already added inside gi/gh by the WMMA stage) --
// hold: fp32 old hidden [16][64]; hnewf: optional fp32 new-h destination;
// d1/d2: bf16 destinations (A buffers for downstream WMMA).
__device__ inline void gru_ew(const float* gi, const float* gh,
                              const float* hold, float* hnewf,
                              unsigned short* d1, int s1,
                              unsigned short* d2, int s2, int tid) {
    for (int idx = tid; idx < 16 * 64; idx += BLOCK) {
        const int row = idx >> 6, j = idx & 63;
        const float ir = gi[row * 192 + j],       hr = gh[row * 192 + j];
        const float iz = gi[row * 192 + 64 + j],  hz = gh[row * 192 + 64 + j];
        const float ig = gi[row * 192 + 128 + j], hg = gh[row * 192 + 128 + j];
        const float r  = sigm(ir + hr);
        const float zt = sigm(iz + hz);
        const float ng = tanh_fast(ig + r * hg);
        const float hn = (1.f - zt) * ng + zt * hold[row * 64 + j];
        if (hnewf) hnewf[row * 64 + j] = hn;
        const unsigned short hb = f2bf(hn);
        d1[row * s1 + j] = hb;
        d2[row * s2 + j] = hb;
    }
}

// --- weight packer: fp32 [N][K] -> bf16 WMMA B-fragments, zero padded -------
__global__ void pack_w(const float* __restrict__ W, unsigned short* __restrict__ dst,
                       int K_orig, int N_orig, int KT, int NT) {
    const int d = blockIdx.x * blockDim.x + threadIdx.x;
    const int total = KT * NT * 512;
    if (d >= total) return;
    const int tile = d >> 9, within = d & 511;
    const int lane = within >> 4, e = within & 15;
    const int v = e >> 1, h = e & 1;
    const int kt = tile % KT, nt = tile / KT;
    const int K = kt * 32 + (e & 7) + ((lane >> 4) << 3) + ((e >> 3) << 4); // B-fragment: elements carry K (table order)
    const int N = nt * 16 + (lane & 15);     // lane carries N
    const float val = (K < K_orig && N < N_orig) ? W[N * K_orig + K] : 0.f;
    dst[d] = f2bf(val);
}

struct KParams {
    const float *z, *F, *H, *m1x0, *hQ0, *hSig0, *hS0;
    const float *bihQ, *bhhQ, *bihSig, *bhhSig, *bihS, *bhhS;
    const float *b1, *b2a, *b2b, *b3, *b4;
    const float *W5, *b5, *W6, *b6, *W7, *b7;
    const unsigned short *pWihQ, *pWhhQ, *pWihSig, *pWhhSig, *pWihS, *pWhhS;
    const unsigned short *pW1, *pW2a, *pW2b, *pW3, *pW4;
    float* out;
};

__global__ __launch_bounds__(BLOCK) void kalman_persistent(KParams p) {
    // fp32 recurrent state + gate buffers
    __shared__ float s_F[64], s_H[64];
    __shared__ float s_xpost[16][8], s_xpost_prev[16][8], s_xprior_prev[16][8];
    __shared__ float s_yprev[16][8], s_xprior[16][8], s_m1y[16][8], s_dy[16][8];
    __shared__ float s_fwev[16][8], s_fwupd[16][8], s_obsd[16][8], s_obsi[16][8];
    __shared__ float s_hQ[16][64], s_hSig[16][64], s_hS[16][64];
    __shared__ float s_gi[16][192], s_gh[16][192];
    __shared__ float s_red[16][64];                 // FC2b split-K reduction
    // bf16 WMMA A-operand buffers (16-byte aligned for ds_load_b128)
    __shared__ alignas(16) unsigned short s_o5A_bf[16][64];    // o5 pad K=64
    __shared__ alignas(16) unsigned short s_hQ_bf[16][64];
    __shared__ alignas(16) unsigned short s_sigA_bf[16][128];  // [hQ | o6] pad
    __shared__ alignas(16) unsigned short s_sA_bf[16][160];    // [o1 | o7] pad
    __shared__ alignas(16) unsigned short s_hSigS_bf[16][128]; // [hSig | hS]
    __shared__ alignas(16) unsigned short s_hSo2_bf[16][128];  // [hS | o2]
    __shared__ alignas(16) unsigned short s_cat4_bf[16][128];  // [hSig | o3]
    __shared__ alignas(16) unsigned short s_hid[16][5120];     // FC2a hidden

    const int tid = threadIdx.x, wave = tid >> 5, lane = tid & 31;
    const int bbase = blockIdx.x * 16;

    // ---- init -------------------------------------------------------------
    for (int i = tid; i < 64; i += BLOCK) { s_F[i] = p.F[i]; s_H[i] = p.H[i]; }
    for (int idx = tid; idx < 128; idx += BLOCK) {
        const int r = idx >> 3, m = idx & 7;
        const float x0 = p.m1x0[(bbase + r) * 8 + m];
        s_xpost[r][m] = x0; s_xpost_prev[r][m] = x0; s_xprior_prev[r][m] = x0;
    }
    for (int idx = tid; idx < 16 * 64; idx += BLOCK) {
        const int r = idx >> 6, j = idx & 63;
        const float hq = p.hQ0[j], hsg = p.hSig0[j], hs = p.hS0[j];
        s_hQ[r][j] = hq;   s_hQ_bf[r][j] = f2bf(hq);
        s_hSig[r][j] = hsg;
        s_hSigS_bf[r][j] = f2bf(hsg);
        s_hSigS_bf[r][64 + j] = f2bf(hs);
        s_hS[r][j] = hs;   s_hSo2_bf[r][j] = f2bf(hs);
        s_o5A_bf[r][j] = 0;
    }
    for (int idx = tid; idx < 16 * 128; idx += BLOCK) s_sigA_bf[idx >> 7][idx & 127] = 0;
    for (int idx = tid; idx < 16 * 160; idx += BLOCK) s_sA_bf[idx / 160][idx % 160] = 0;
    __syncthreads();
    for (int idx = tid; idx < 128; idx += BLOCK) {       // y_prev0 = m1x0 @ H.T
        const int r = idx >> 3, n = idx & 7;
        float a = 0.f;
        for (int k = 0; k < 8; ++k) a += s_xpost[r][k] * s_H[n * 8 + k];
        s_yprev[r][n] = a;
    }
    __syncthreads();

    // ---- time recurrence ----------------------------------------------------
    for (int t = 0; t < TSTEPS; ++t) {
        // x_prior = x_post @ F.T
        for (int idx = tid; idx < 128; idx += BLOCK) {
            const int r = idx >> 3, m = idx & 7;
            float a = 0.f;
            for (int k = 0; k < 8; ++k) a += s_xpost[r][k] * s_F[m * 8 + k];
            s_xprior[r][m] = a;
        }
        __syncthreads();
        // m1y = x_prior @ H.T ; dy = yt - m1y
        for (int idx = tid; idx < 128; idx += BLOCK) {
            const int r = idx >> 3, n = idx & 7;
            float a = 0.f;
            for (int k = 0; k < 8; ++k) a += s_xprior[r][k] * s_H[n * 8 + k];
            s_m1y[r][n] = a;
            const float y = p.z[(((size_t)(bbase + r)) * TSTEPS + t) * 8 + n];
            s_dy[r][n] = y - a;
        }
        __syncthreads();
        // four L2-normalized feature vectors per batch row
        for (int idx = tid; idx < 64; idx += BLOCK) {
            const int r = idx >> 2, v = idx & 3;
            float tmp[8]; float ss = 0.f;
            for (int k = 0; k < 8; ++k) {
                const float y = p.z[(((size_t)(bbase + r)) * TSTEPS + t) * 8 + k];
                float d;
                if (v == 0)      d = y - s_yprev[r][k];
                else if (v == 1) d = y - s_m1y[r][k];
                else if (v == 2) d = s_xpost[r][k] - s_xpost_prev[r][k];
                else             d = s_xpost[r][k] - s_xprior_prev[r][k];
                tmp[k] = d; ss += d * d;
            }
            const float inv = __fdividef(1.f, fmaxf(sqrtf(ss), 1e-12f));
            float* dst = (v == 0) ? s_obsd[r] : (v == 1) ? s_obsi[r]
                       : (v == 2) ? s_fwev[r] : s_fwupd[r];
            for (int k = 0; k < 8; ++k) dst[k] = tmp[k] * inv;
        }
        __syncthreads();
        // tiny FCs (K=8/16) on VALU: o5, o6, o7 -> bf16 A buffers
        for (int idx = tid; idx < 16 * 40; idx += BLOCK) {
            const int r = idx / 40, c = idx % 40;
            float a = p.b5[c];
            for (int k = 0; k < 8; ++k) a += s_fwev[r][k] * p.W5[c * 8 + k];
            s_o5A_bf[r][c] = f2bf(fmaxf(a, 0.f));
        }
        for (int idx = tid; idx < 16 * 40; idx += BLOCK) {
            const int r = idx / 40, c = idx % 40;
            float a = p.b6[c];
            for (int k = 0; k < 8; ++k) a += s_fwupd[r][k] * p.W6[c * 8 + k];
            s_sigA_bf[r][64 + c] = f2bf(fmaxf(a, 0.f));
        }
        for (int idx = tid; idx < 16 * 80; idx += BLOCK) {
            const int r = idx / 80, c = idx % 80;
            float a = p.b7[c];
            for (int k = 0; k < 8; ++k) a += s_obsd[r][k] * p.W7[c * 16 + k];
            for (int k = 0; k < 8; ++k) a += s_obsi[r][k] * p.W7[c * 16 + 8 + k];
            s_sA_bf[r][64 + c] = f2bf(fmaxf(a, 0.f));
        }
        __syncthreads();
        // GRU_Q (K=40->64 / 64, N=192)
        wmma_fc<2, false>(&s_o5A_bf[0][0], 64, p.pWihQ, p.bihQ, &s_gi[0][0], 192, 12, wave, lane);
        wmma_fc<2, false>(&s_hQ_bf[0][0], 64, p.pWhhQ, p.bhhQ, &s_gh[0][0], 192, 12, wave, lane);
        __syncthreads();
        gru_ew(&s_gi[0][0], &s_gh[0][0], &s_hQ[0][0], &s_hQ[0][0],
               &s_hQ_bf[0][0], 64, &s_sigA_bf[0][0], 128, tid);
        __syncthreads();
        // GRU_Sigma (K=104->128 / 64, N=192); old h = carry (o4 of prev step)
        wmma_fc<4, false>(&s_sigA_bf[0][0], 128, p.pWihSig, p.bihSig, &s_gi[0][0], 192, 12, wave, lane);
        wmma_fc<2, false>(&s_hSigS_bf[0][0], 128, p.pWhhSig, p.bhhSig, &s_gh[0][0], 192, 12, wave, lane);
        __syncthreads();
        // hSig_t used only this step (carry becomes o4) -> bf16 dests only
        gru_ew(&s_gi[0][0], &s_gh[0][0], &s_hSig[0][0], nullptr,
               &s_hSigS_bf[0][0], 128, &s_cat4_bf[0][0], 128, tid);
        __syncthreads();
        // o1 = relu(hSig @ W1.T + b1) -> s_sA_bf[:, 0:64]
        wmma_fc_bf16out<2, true>(&s_hSigS_bf[0][0], 128, p.pW1, p.b1,
                                 &s_sA_bf[0][0], 160, 4, wave, lane);
        __syncthreads();
        // GRU_S (K=144->160 / 64, N=192); gh reads old hS from s_hSo2_bf[:,0:64]
        wmma_fc<5, false>(&s_sA_bf[0][0], 160, p.pWihS, p.bihS, &s_gi[0][0], 192, 12, wave, lane);
        wmma_fc<2, false>(&s_hSo2_bf[0][0], 128, p.pWhhS, p.bhhS, &s_gh[0][0], 192, 12, wave, lane);
        __syncthreads();
        gru_ew(&s_gi[0][0], &s_gh[0][0], &s_hS[0][0], &s_hS[0][0],
               &s_hSigS_bf[0][64], 128, &s_hSo2_bf[0][0], 128, tid);
        for (int idx = tid; idx < 1024; idx += BLOCK) (&s_red[0][0])[idx] = 0.f;
        __syncthreads();
        // FC2a: [hSig|hS](16x128) -> 16x5120 bf16 hidden (relu)
        wmma_fc_bf16out<4, true>(&s_hSigS_bf[0][0], 128, p.pW2a, p.b2a,
                                 &s_hid[0][0], 5120, 320, wave, lane);
        __syncthreads();
        // FC2b split-K: 16 waves = 4 N-tiles x 4 K-chunks of 40 tiles
        {
            const int nt = wave & 3, chunk = wave >> 2;
            v8f acc = {};
            const unsigned short* bp =
                p.pW2b + (((size_t)(nt * 160 + chunk * 40)) << 9) + (lane << 4);
            for (int kt = 0; kt < 40; ++kt) {
                v16bf a = a_frag_bf(&s_hid[0][0], 5120, (chunk * 40 + kt) * 32, lane);
                v16bf b = __builtin_bit_cast(v16bf, *(const v16u*)bp);
                bp += 512;
                acc = __builtin_amdgcn_wmma_f32_16x16x32_bf16(false, a, false, b,
                                                              (short)0, acc, false, false);
            }
            const int col = (nt << 4) + (lane & 15);
#pragma unroll
            for (int i = 0; i < 8; ++i)
                atomicAdd(&s_red[i + ((lane >> 4) << 3)][col], acc[i]);
        }
        __syncthreads();
        // o2 = red + b2b -> bf16 into s_hSo2_bf[:, 64:128] (fp32 recomputed later)
        for (int idx = tid; idx < 1024; idx += BLOCK) {
            const int r = idx >> 6, j = idx & 63;
            s_hSo2_bf[r][64 + j] = f2bf(s_red[r][j] + p.b2b[j]);
        }
        __syncthreads();
        // o3 = relu([hS|o2] @ W3.T + b3) -> s_cat4_bf[:, 64:128]
        wmma_fc_bf16out<4, true>(&s_hSo2_bf[0][0], 128, p.pW3, p.b3,
                                 &s_cat4_bf[0][64], 128, 4, wave, lane);
        __syncthreads();
        // o4 = relu([hSig|o3] @ W4.T + b4) -> new hSig carry (fp32 + bf16)
        wmma_fc_dual<4>(&s_cat4_bf[0][0], 128, p.pW4, p.b4,
                        &s_hSig[0][0], 64, &s_hSigS_bf[0][0], 128, 4, wave, lane);
        __syncthreads();
        // correction + state rotation + output (KG row-major o2 in fp32)
        for (int idx = tid; idx < 128; idx += BLOCK) {
            const int r = idx >> 3, m = idx & 7;
            float xn = s_xprior[r][m];
            for (int n2 = 0; n2 < 8; ++n2)
                xn += (s_red[r][m * 8 + n2] + p.b2b[m * 8 + n2]) * s_dy[r][n2];
            p.out[(((size_t)(bbase + r)) * TSTEPS + t) * 8 + m] = xn;
            const float oldpost = s_xpost[r][m];
            s_xpost_prev[r][m]  = oldpost;
            s_xprior_prev[r][m] = s_xprior[r][m];
            s_xpost[r][m]       = xn;
            s_yprev[r][m]       = p.z[(((size_t)(bbase + r)) * TSTEPS + t) * 8 + m];
        }
        __syncthreads();
    }
}

extern "C" void kernel_launch(void* const* d_in, const int* in_sizes, int n_in,
                              void* d_out, int out_size, void* d_ws, size_t ws_size,
                              hipStream_t stream) {
    (void)in_sizes; (void)n_in; (void)out_size; (void)ws_size;
    const float* z      = (const float*)d_in[0];
    const float* F      = (const float*)d_in[1];
    const float* H      = (const float*)d_in[2];
    const float* m1x0   = (const float*)d_in[3];
    const float* hQ0    = (const float*)d_in[4];
    const float* hSig0  = (const float*)d_in[5];
    const float* hS0    = (const float*)d_in[6];
    const float* WihQ   = (const float*)d_in[7];
    const float* WhhQ   = (const float*)d_in[8];
    const float* bihQ   = (const float*)d_in[9];
    const float* bhhQ   = (const float*)d_in[10];
    const float* WihSig = (const float*)d_in[11];
    const float* WhhSig = (const float*)d_in[12];
    const float* bihSig = (const float*)d_in[13];
    const float* bhhSig = (const float*)d_in[14];
    const float* WihS   = (const float*)d_in[15];
    const float* WhhS   = (const float*)d_in[16];
    const float* bihS   = (const float*)d_in[17];
    const float* bhhS   = (const float*)d_in[18];
    const float* W1     = (const float*)d_in[19];
    const float* b1     = (const float*)d_in[20];
    const float* W2a    = (const float*)d_in[21];
    const float* b2a    = (const float*)d_in[22];
    const float* W2b    = (const float*)d_in[23];
    const float* b2b    = (const float*)d_in[24];
    const float* W3     = (const float*)d_in[25];
    const float* b3     = (const float*)d_in[26];
    const float* W4     = (const float*)d_in[27];
    const float* b4     = (const float*)d_in[28];
    const float* W5     = (const float*)d_in[29];
    const float* b5     = (const float*)d_in[30];
    const float* W6     = (const float*)d_in[31];
    const float* b6     = (const float*)d_in[32];
    const float* W7     = (const float*)d_in[33];
    const float* b7     = (const float*)d_in[34];

    unsigned short* ws = (unsigned short*)d_ws;
    size_t off = 0;
    auto alloc = [&](int KT, int NT) {
        unsigned short* q = ws + off;
        off += (size_t)KT * NT * 512;
        return q;
    };
    auto pack = [&](const float* W, unsigned short* dst, int K, int N, int KT, int NT) {
        const int total = KT * NT * 512;
        pack_w<<<(total + 255) / 256, 256, 0, stream>>>(W, dst, K, N, KT, NT);
    };

    unsigned short* pWihQ   = alloc(2, 12);   pack(WihQ,   pWihQ,   40,  192, 2, 12);
    unsigned short* pWhhQ   = alloc(2, 12);   pack(WhhQ,   pWhhQ,   64,  192, 2, 12);
    unsigned short* pWihSig = alloc(4, 12);   pack(WihSig, pWihSig, 104, 192, 4, 12);
    unsigned short* pWhhSig = alloc(2, 12);   pack(WhhSig, pWhhSig, 64,  192, 2, 12);
    unsigned short* pWihS   = alloc(5, 12);   pack(WihS,   pWihS,   144, 192, 5, 12);
    unsigned short* pWhhS   = alloc(2, 12);   pack(WhhS,   pWhhS,   64,  192, 2, 12);
    unsigned short* pW1     = alloc(2, 4);    pack(W1,     pW1,     64,  64,  2, 4);
    unsigned short* pW2a    = alloc(4, 320);  pack(W2a,    pW2a,    128, 5120, 4, 320);
    unsigned short* pW2b    = alloc(160, 4);  pack(W2b,    pW2b,    5120, 64, 160, 4);
    unsigned short* pW3     = alloc(4, 4);    pack(W3,     pW3,     128, 64,  4, 4);
    unsigned short* pW4     = alloc(4, 4);    pack(W4,     pW4,     128, 64,  4, 4);

    KParams kp;
    kp.z = z; kp.F = F; kp.H = H; kp.m1x0 = m1x0;
    kp.hQ0 = hQ0; kp.hSig0 = hSig0; kp.hS0 = hS0;
    kp.bihQ = bihQ; kp.bhhQ = bhhQ; kp.bihSig = bihSig; kp.bhhSig = bhhSig;
    kp.bihS = bihS; kp.bhhS = bhhS;
    kp.b1 = b1; kp.b2a = b2a; kp.b2b = b2b; kp.b3 = b3; kp.b4 = b4;
    kp.W5 = W5; kp.b5 = b5; kp.W6 = W6; kp.b6 = b6; kp.W7 = W7; kp.b7 = b7;
    kp.pWihQ = pWihQ; kp.pWhhQ = pWhhQ; kp.pWihSig = pWihSig; kp.pWhhSig = pWhhSig;
    kp.pWihS = pWihS; kp.pWhhS = pWhhS;
    kp.pW1 = pW1; kp.pW2a = pW2a; kp.pW2b = pW2b; kp.pW3 = pW3; kp.pW4 = pW4;
    kp.out = (float*)d_out;

    kalman_persistent<<<dim3(8), dim3(BLOCK), 0, stream>>>(kp);
}
